// VoxelToCostNet_79731772883614
// MI455X (gfx1250) — matrix operand order repaired and checked by reading, not compiled, not run on hardware
//
#include <hip/hip_runtime.h>
#include <hip/hip_bf16.h>

typedef __attribute__((ext_vector_type(16))) __bf16 v16bf;
typedef __attribute__((ext_vector_type(8)))  float  v8f;

#define NXv 256
#define NYv 256
#define NZv 64
#define MPTS 400000

__device__ __forceinline__ unsigned f2bf1(float f) {
  unsigned u = __float_as_uint(f);
  return (u + 0x7FFFu + ((u >> 16) & 1u)) >> 16;   // round-to-nearest-even bf16
}
__device__ __forceinline__ unsigned pkbf(float a, float b) {
  return f2bf1(a) | (f2bf1(b) << 16);
}

union ABfrag {
  unsigned u[8];
  uint4    q[2];
  v16bf    v;
};

__device__ __forceinline__ v8f vzero8() {
  v8f z;
#pragma unroll
  for (int i = 0; i < 8; ++i) z[i] = 0.f;
  return z;
}

__device__ __forceinline__ v8f wmma_bf16_acc(const ABfrag& a, const ABfrag& b, v8f c) {
  // D(f32 16x16) = A(bf16 16x32) * B(bf16 32x16) + C
  return __builtin_amdgcn_wmma_f32_16x16x32_bf16(false, a.v, false, b.v, (short)0, c,
                                                 false, false);
}

// Build a 16x32 bf16 A-fragment from one gathered f32 row (row stride cs floats).
// lg = lane>>4. half=true => K 16..31 are zero (Cin<=16 sources).
__device__ __forceinline__ ABfrag load_a_f32(const float* __restrict__ src, int cs, int g,
                                             int kbase, int lg, bool half) {
  ABfrag a;
  if (g >= 0) {
    const float* fp = src + (size_t)g * cs + kbase + lg * 8;
    float4 f0 = *(const float4*)fp;
    float4 f1 = *(const float4*)(fp + 4);
    a.u[0] = pkbf(f0.x, f0.y); a.u[1] = pkbf(f0.z, f0.w);
    a.u[2] = pkbf(f1.x, f1.y); a.u[3] = pkbf(f1.z, f1.w);
    if (!half) {
      const float* fp2 = fp + 16;
      float4 f2 = *(const float4*)fp2;
      float4 f3 = *(const float4*)(fp2 + 4);
      a.u[4] = pkbf(f2.x, f2.y); a.u[5] = pkbf(f2.z, f2.w);
      a.u[6] = pkbf(f3.x, f3.y); a.u[7] = pkbf(f3.z, f3.w);
    } else {
      a.u[4] = a.u[5] = a.u[6] = a.u[7] = 0u;
    }
  } else {
#pragma unroll
    for (int i = 0; i < 8; ++i) a.u[i] = 0u;
  }
  return a;
}

// ---------------------------------------------------------------------------
// Sparse submanifold 3D conv: one wave per 16-point tile, 27 offsets,
// KC K-chunks of 32, 4 Cout tiles of 16 (Cout=64). Output rows are bf16[64].
// B-fragments for all 4 Cout tiles are batch-loaded before the 4 WMMAs so the
// scheduler can overlap the L2 loads with matrix issue.
// ---------------------------------------------------------------------------
template <int KC, bool SRCF32, bool RELU>
__global__ __launch_bounds__(256) void sparse_conv_kernel(
    const void* __restrict__ src, const int* __restrict__ indices,
    const int* __restrict__ table, const unsigned* __restrict__ wp,
    const float* __restrict__ bias, unsigned short* __restrict__ out, int ntiles) {
  const int lane = threadIdx.x & 31;
  const int wave = threadIdx.x >> 5;
  const int tile = blockIdx.x * (blockDim.x >> 5) + wave;
  if (tile >= ntiles) return;
  const int lg = lane >> 4;
  const int r  = lane & 15;
  const int p  = tile * 16 + r;

  const int x = indices[p * 4 + 1];
  const int y = indices[p * 4 + 2];
  const int z = indices[p * 4 + 3];

  v8f acc[4];
#pragma unroll
  for (int i = 0; i < 4; ++i) acc[i] = vzero8();

  for (int off = 0; off < 27; ++off) {
    const int dx = off / 9 - 1, dy = (off / 3) % 3 - 1, dz = off % 3 - 1;
    const int nx = x + dx, ny = y + dy, nz = z + dz;
    int g = -1;
    if (((unsigned)nx < (unsigned)NXv) & ((unsigned)ny < (unsigned)NYv) &
        ((unsigned)nz < (unsigned)NZv)) {
      const int flat = nx * (NYv * NZv) + ny * NZv + nz;
      g = table[flat] - 1;  // -1 => empty voxel
    }
#pragma unroll
    for (int kc = 0; kc < KC; ++kc) {
      ABfrag a;
      if (SRCF32) {
        a = load_a_f32((const float*)src, 16, g, kc * 32, lg, true);  // Cin=16
      } else {
        if (g >= 0) {
          const char* bp = (const char*)src + (size_t)g * 128 + (size_t)(kc * 32 + lg * 8) * 2;
          a.q[0] = *(const uint4*)bp;
          a.q[1] = *(const uint4*)(bp + 32);
        } else {
#pragma unroll
          for (int i = 0; i < 8; ++i) a.u[i] = 0u;
        }
      }
      const unsigned* bbase = wp + (((size_t)off * KC + kc) * 4 * 32 + lane) * 8;
      ABfrag bf[4];
#pragma unroll
      for (int ct = 0; ct < 4; ++ct) {        // batch all B loads (one clause)
        bf[ct].q[0] = *(const uint4*)(bbase + ct * 1024);
        bf[ct].q[1] = *(const uint4*)(bbase + ct * 1024 + 4);
      }
#pragma unroll
      for (int ct = 0; ct < 4; ++ct) {        // then back-to-back WMMAs
        acc[ct] = wmma_bf16_acc(a, bf[ct], acc[ct]);
      }
    }
  }

  // C layout: lane<16 -> N=lane, M=j; lane>=16 -> N=lane-16, M=j+8
  const int n0 = r;
#pragma unroll
  for (int ct = 0; ct < 4; ++ct) {
    const float bs = bias[ct * 16 + n0];
#pragma unroll
    for (int j = 0; j < 8; ++j) {
      const int m = lg ? (j + 8) : j;
      float v = acc[ct][j] + bs;
      if (RELU) v = fmaxf(v, 0.f);
      out[(size_t)(tile * 16 + m) * 64 + ct * 16 + n0] = (unsigned short)f2bf1(v);
    }
  }
}

// ---------------------------------------------------------------------------
// Dense 5x5 conv (HWC f32), one wave per 16-pixel tile, optional fused
// 1x1-projection skip input; always ends with ReLU (matches resblock fusion).
// ---------------------------------------------------------------------------
template <int KC0, bool HALF0, int SKC, int CT>
__global__ __launch_bounds__(256) void dense_conv_kernel(
    const float* __restrict__ src0, int cs0, const unsigned* __restrict__ w0,
    const float* __restrict__ src1, int cs1, const unsigned* __restrict__ w1p,
    const float* __restrict__ b0, const float* __restrict__ b1,
    float* __restrict__ out, int ost, int outch) {
  const int lane = threadIdx.x & 31;
  const int wave = threadIdx.x >> 5;
  const int tile = blockIdx.x * (blockDim.x >> 5) + wave;  // 4096 tiles total
  if (tile >= 4096) return;
  const int lg  = lane >> 4;
  const int r   = lane & 15;
  const int py  = tile >> 4;          // image row (x in reference)
  const int px0 = (tile & 15) << 4;   // start column (y in reference)
  const int px  = px0 + r;

  v8f acc[CT];
#pragma unroll
  for (int i = 0; i < CT; ++i) acc[i] = vzero8();

  for (int off = 0; off < 25; ++off) {
    const int dy = off / 5 - 2, dx = off % 5 - 2;
    const int qy = py + dy, qx = px + dx;
    const int g = (((unsigned)qy < 256u) && ((unsigned)qx < 256u)) ? (qy * 256 + qx) : -1;
#pragma unroll
    for (int kc = 0; kc < KC0; ++kc) {
      ABfrag a = load_a_f32(src0, cs0, g, kc * 32, lg, HALF0);
      const unsigned* bbase = w0 + (((size_t)off * KC0 + kc) * CT * 32 + lane) * 8;
      ABfrag bf[CT];
#pragma unroll
      for (int ct = 0; ct < CT; ++ct) {
        bf[ct].q[0] = *(const uint4*)(bbase + ct * 1024);
        bf[ct].q[1] = *(const uint4*)(bbase + ct * 1024 + 4);
      }
#pragma unroll
      for (int ct = 0; ct < CT; ++ct) {
        acc[ct] = wmma_bf16_acc(a, bf[ct], acc[ct]);
      }
    }
  }
  if (SKC > 0) {  // fused 1x1 projection on src1 (residual path)
    const int g = py * 256 + px;
#pragma unroll
    for (int kc = 0; kc < SKC; ++kc) {
      ABfrag a = load_a_f32(src1, cs1, g, kc * 32, lg, false);
      const unsigned* bbase = w1p + ((size_t)kc * CT * 32 + lane) * 8;
      ABfrag bf[CT];
#pragma unroll
      for (int ct = 0; ct < CT; ++ct) {
        bf[ct].q[0] = *(const uint4*)(bbase + ct * 1024);
        bf[ct].q[1] = *(const uint4*)(bbase + ct * 1024 + 4);
      }
#pragma unroll
      for (int ct = 0; ct < CT; ++ct) {
        acc[ct] = wmma_bf16_acc(a, bf[ct], acc[ct]);
      }
    }
  }

  const int n0 = r;
#pragma unroll
  for (int ct = 0; ct < CT; ++ct) {
    const int c = ct * 16 + n0;
    if (c < outch) {
      float bs = b0[c];
      if (SKC > 0) bs += b1[c];
#pragma unroll
      for (int j = 0; j < 8; ++j) {
        const int m = lg ? (j + 8) : j;
        float v = acc[ct][j] + bs;
        v = fmaxf(v, 0.f);
        out[(size_t)(py * 256 + px0 + m) * ost + c] = v;
      }
    }
  }
}

// ---------------------------------------------------------------------------
// Helpers: weight repack (f32 -> per-lane WMMA B-fragment bf16 layout),
// fill, segment-max scatter, -inf fixup.
// ---------------------------------------------------------------------------
__device__ __forceinline__ float fetch_w(const float* w, int mode, int Cin, int Cout,
                                         int noff, int off, int k, int n) {
  if (k >= Cin || n >= Cout) return 0.f;
  if (mode == 0) return w[((size_t)off * Cin + k) * Cout + n];        // (noff,Cin,Cout)
  if (mode == 1) return w[((size_t)n * Cin + k) * noff + off];        // (Cout,Cin,5,5)
  return w[(size_t)n * Cin + k];                                      // (Cout,Cin,1,1)
}

__global__ void repack_kernel(const float* __restrict__ w, unsigned* __restrict__ dst,
                              int mode, int noff, int Cin, int Cout, int KC, int CT) {
  const int total = noff * KC * CT * 256;
  const int idx = blockIdx.x * blockDim.x + threadIdx.x;
  if (idx >= total) return;
  const int j    = idx & 7;
  const int lane = (idx >> 3) & 31;
  int rest = idx >> 8;
  const int ct = rest % CT; rest /= CT;
  const int kc = rest % KC; rest /= KC;
  const int off = rest;
  const int lg = lane >> 4, n0 = lane & 15;
  const int n  = ct * 16 + n0;
  const int k0 = kc * 32 + lg * 8 + (j >> 2) * 16 + 2 * (j & 3);  // B layout: v_j holds rows k0 (lo), k0+1 (hi)
  const float f0 = fetch_w(w, mode, Cin, Cout, noff, off, k0, n);
  const float f1 = fetch_w(w, mode, Cin, Cout, noff, off, k0 + 1, n);
  dst[idx] = pkbf(f0, f1);
}

__global__ void fill_u32_kernel(unsigned* __restrict__ p, unsigned v, int n) {
  const int i = blockIdx.x * blockDim.x + threadIdx.x;
  if (i < n) p[i] = v;
}

__global__ void segmax_kernel(const unsigned short* __restrict__ h,
                              const int* __restrict__ indices, float* __restrict__ bev) {
  const int i = blockIdx.x * blockDim.x + threadIdx.x;
  if (i >= MPTS * 64) return;
  const int p = i >> 6, c = i & 63;
  const float v = __uint_as_float(((unsigned)h[i]) << 16);
  const int x = indices[p * 4 + 1], y = indices[p * 4 + 2];
  float* addr = bev + ((size_t)(x * 256 + y)) * 64 + c;
  // order-independent fp32 max via sign-split integer atomics
  if (v >= 0.f) atomicMax((int*)addr, __float_as_int(v));
  else          atomicMin((unsigned int*)addr, __float_as_uint(v));
}

__global__ void fixup_neginf_kernel(float* __restrict__ bev, int n) {
  const int i = blockIdx.x * blockDim.x + threadIdx.x;
  if (i < n && __float_as_uint(bev[i]) == 0xFF800000u) bev[i] = 0.f;
}

// ---------------------------------------------------------------------------
extern "C" void kernel_launch(void* const* d_in, const int* in_sizes, int n_in,
                              void* d_out, int out_size, void* d_ws, size_t ws_size,
                              hipStream_t stream) {
  (void)in_sizes; (void)n_in; (void)out_size; (void)ws_size;
  const float* features = (const float*)d_in[0];
  const float* w1   = (const float*)d_in[1];
  const float* b1   = (const float*)d_in[2];
  const float* w2   = (const float*)d_in[3];
  const float* b2   = (const float*)d_in[4];
  const float* w3   = (const float*)d_in[5];
  const float* b3   = (const float*)d_in[6];
  const float* cw11 = (const float*)d_in[7];
  const float* cb11 = (const float*)d_in[8];
  const float* cw12 = (const float*)d_in[9];
  const float* cb12 = (const float*)d_in[10];
  const float* cpw1 = (const float*)d_in[11];
  const float* cpb1 = (const float*)d_in[12];
  const float* cw21 = (const float*)d_in[13];
  const float* cb21 = (const float*)d_in[14];
  const float* cw22 = (const float*)d_in[15];
  const float* cb22 = (const float*)d_in[16];
  const float* cpw2 = (const float*)d_in[17];
  const float* cpb2 = (const float*)d_in[18];
  const int* indices = (const int*)d_in[19];
  const int* table   = (const int*)d_in[20];
  float* outp = (float*)d_out;

  char* ws = (char*)d_ws;
  const size_t OFF_H1  = 0;                       // M*64 bf16
  const size_t OFF_H2  = OFF_H1 + 51200000;       // M*64 bf16
  const size_t OFF_BEV = OFF_H2 + 51200000;       // 256*256*64 f32
  const size_t OFF_T1  = OFF_BEV + 16777216;      // 256*256*64 f32
  const size_t OFF_R1  = OFF_T1 + 16777216;       // 256*256*64 f32
  const size_t OFF_T2  = OFF_R1 + 16777216;       // 256*256*16 f32 (3ch padded)
  const size_t OFF_PW1  = OFF_T2 + 4194304;
  const size_t OFF_PW2  = OFF_PW1 + 110592;
  const size_t OFF_PW3  = OFF_PW2 + 221184;
  const size_t OFF_PC11 = OFF_PW3 + 221184;
  const size_t OFF_PC12 = OFF_PC11 + 204800;
  const size_t OFF_PP1  = OFF_PC12 + 204800;
  const size_t OFF_PC21 = OFF_PP1 + 8192;
  const size_t OFF_PC22 = OFF_PC21 + 51200;
  const size_t OFF_PP2  = OFF_PC22 + 25600;

  unsigned short* h1 = (unsigned short*)(ws + OFF_H1);
  unsigned short* h2 = (unsigned short*)(ws + OFF_H2);
  float* bev = (float*)(ws + OFF_BEV);
  float* t1  = (float*)(ws + OFF_T1);
  float* r1  = (float*)(ws + OFF_R1);
  float* t2  = (float*)(ws + OFF_T2);
  unsigned* pw1  = (unsigned*)(ws + OFF_PW1);
  unsigned* pw2  = (unsigned*)(ws + OFF_PW2);
  unsigned* pw3  = (unsigned*)(ws + OFF_PW3);
  unsigned* pc11 = (unsigned*)(ws + OFF_PC11);
  unsigned* pc12 = (unsigned*)(ws + OFF_PC12);
  unsigned* pp1  = (unsigned*)(ws + OFF_PP1);
  unsigned* pc21 = (unsigned*)(ws + OFF_PC21);
  unsigned* pc22 = (unsigned*)(ws + OFF_PC22);
  unsigned* pp2  = (unsigned*)(ws + OFF_PP2);

  // --- repack all weights into WMMA B-fragment bf16 layout (L2-resident) ---
  auto rp = [&](const float* w, unsigned* dst, int mode, int noff, int Cin, int Cout,
                int KC, int CT) {
    const int total = noff * KC * CT * 256;
    repack_kernel<<<(total + 255) / 256, 256, 0, stream>>>(w, dst, mode, noff, Cin, Cout, KC, CT);
  };
  rp(w1,   pw1,  0, 27, 16, 64, 1, 4);
  rp(w2,   pw2,  0, 27, 64, 64, 2, 4);
  rp(w3,   pw3,  0, 27, 64, 64, 2, 4);
  rp(cw11, pc11, 1, 25, 64, 64, 2, 4);
  rp(cw12, pc12, 1, 25, 64, 64, 2, 4);
  rp(cpw1, pp1,  2, 1,  64, 64, 2, 4);
  rp(cw21, pc21, 1, 25, 64, 3,  2, 1);
  rp(cw22, pc22, 1, 25, 3,  3,  1, 1);
  rp(cpw2, pp2,  2, 1,  64, 3,  2, 1);

  // --- sparse 3D conv stack (25000 tiles of 16 points; 8 waves/block) ---
  sparse_conv_kernel<1, true,  true ><<<3125, 256, 0, stream>>>(features, indices, table, pw1, b1, h1, 25000);
  sparse_conv_kernel<2, false, true ><<<3125, 256, 0, stream>>>(h1,       indices, table, pw2, b2, h2, 25000);
  sparse_conv_kernel<2, false, false><<<3125, 256, 0, stream>>>(h2,       indices, table, pw3, b3, h1, 25000);

  // --- BEV segment-max ---
  fill_u32_kernel<<<16384, 256, 0, stream>>>((unsigned*)bev, 0xFF800000u, 256 * 256 * 64);
  segmax_kernel<<<(MPTS * 64 + 255) / 256, 256, 0, stream>>>(h1, indices, bev);
  fixup_neginf_kernel<<<16384, 256, 0, stream>>>(bev, 256 * 256 * 64);

  // --- dense resblocks (4096 pixel tiles; 8 waves/block) ---
  fill_u32_kernel<<<4096, 256, 0, stream>>>((unsigned*)t2, 0u, 256 * 256 * 16);

  // resblock1: t1 = relu(conv5x5(bev)); r1 = relu(conv5x5(t1) + conv1x1(bev))
  dense_conv_kernel<2, false, 0, 4><<<512, 256, 0, stream>>>(bev, 64, pc11, nullptr, 0, nullptr,
                                                             cb11, nullptr, t1, 64, 64);
  dense_conv_kernel<2, false, 2, 4><<<512, 256, 0, stream>>>(t1, 64, pc12, bev, 64, pp1,
                                                             cb12, cpb1, r1, 64, 64);
  // resblock2: t2 = relu(conv5x5(r1)); out = relu(conv5x5(t2) + conv1x1(r1))
  dense_conv_kernel<2, false, 0, 1><<<512, 256, 0, stream>>>(r1, 64, pc21, nullptr, 0, nullptr,
                                                             cb21, nullptr, t2, 16, 3);
  dense_conv_kernel<1, true,  2, 1><<<512, 256, 0, stream>>>(t2, 16, pc22, r1, 64, pp2,
                                                             cb22, cpb2, outp, 3, 3);
}
